// ProteinBertCrossAttention_32950989095468
// MI455X (gfx1250) — matrix operand order repaired and checked by reading, not compiled, not run on hardware
//
#include <hip/hip_runtime.h>

typedef __attribute__((ext_vector_type(16))) _Float16 v16h;
typedef __attribute__((ext_vector_type(8)))  float    v8f;

#define BATCH   8
#define SEQ     8192
#define DT      1024          // DIM_TOKEN (GEMM K)
#define DG      512           // DIM_GLOBAL = HEADS*DIM_HEAD (GEMM N)
#define NHEADS  8
#define MB      128           // rows per block
#define ROWT    8             // MB/16 row-tiles
#define NT      32            // 512/16 col-tiles per matrix
#define KT      32            // 1024/32 k-tiles

// workspace layout (bytes)
#define WS_QS    0            // 8*512 f32      = 16384
#define WS_GNUM  16384        // 8*512 f32      = 16384
#define WS_GDEN  32768        // 64 f32         = 256
#define WS_WB    36864        // 2*32*32*512 f16 = 2 MiB (32B aligned)

// Branch-free tanh: clamp, t = exp(2x), tanh = (t-1) * rcp(t+1).  v_exp + v_rcp.
__device__ __forceinline__ float fast_tanh(float x) {
    float xc = fminf(fmaxf(x, -9.0f), 9.0f);     // tanh(9) == 1.0f in fp32
    float e  = __expf(2.0f * xc);                // <= exp(18), finite
    return (e - 1.0f) * __builtin_amdgcn_rcpf(e + 1.0f);
}

// Branch-free exact-GELU via Abramowitz-Stegun 7.1.26 erf (|err| < 1.5e-7).
__device__ __forceinline__ float gelu_erf(float x) {
    float z  = x * 0.70710678118654752f;
    float az = fabsf(z);
    float t  = __builtin_amdgcn_rcpf(fmaf(0.3275911f, az, 1.0f));   // denom >= 1
    float p  = t * (0.254829592f +
               t * (-0.284496736f +
               t * (1.421413741f +
               t * (-1.453152027f +
               t * 1.061405429f))));
    float erfabs = 1.0f - p * __expf(-az * az);
    float erfv   = copysignf(erfabs, z);
    return 0.5f * x * (1.0f + erfv);
}

__global__ void zero_kernel(float* __restrict__ gnum, float* __restrict__ gden) {
    int idx = blockIdx.x * 256 + threadIdx.x;
    if (idx < BATCH * DG) gnum[idx] = 0.0f;
    if (idx < BATCH * NHEADS) gden[idx] = 0.0f;
}

// qs[b][h*64+d] = tanh(annotation[b] @ Wq)[h*64+d] * DIM_HEAD^-0.5   (exact fp32)
__global__ void qs_kernel(const float* __restrict__ ann, const float* __restrict__ Wq,
                          float* __restrict__ qs) {
    int idx = blockIdx.x * 256 + threadIdx.x;       // 0..4095
    int b = idx >> 9, col = idx & (DG - 1);
    const float* a = ann + b * DG;
    float acc = 0.0f;
    for (int i = 0; i < DG; ++i) acc = fmaf(a[i], Wq[i * DG + col], acc);
    qs[idx] = tanhf(acc) * 0.125f;                  // scale = 64^-0.5 folded in
}

// Pre-swizzle Wk/Wv (fp32 row-major [K=1024][N=512]) into WMMA-B fragment layout, f16.
// Fragment (m,ct,kt): 32 lanes x 16 halves contiguous.  B-matrix 32x16 layout:
// lanes 0-15: N=lane, halves x<8 hold K=kt*32+x, x>=8 hold K=kt*32+8+x ; lanes 16-31: same +8.
__global__ void swz_kernel(const float* __restrict__ Wk, const float* __restrict__ Wv,
                           _Float16* __restrict__ wb) {
    int idx = blockIdx.x * 256 + threadIdx.x;       // 0 .. 2^20-1
    int x  =  idx        & 15;
    int l  = (idx >> 4)  & 31;
    int kt = (idx >> 9)  & 31;
    int ct = (idx >> 14) & 31;
    int m  =  idx >> 19;
    int K = kt * 32 + ((x < 8) ? x : x + 8) + ((l < 16) ? 0 : 8);
    int n = ct * 16 + (l & 15);
    const float* W = m ? Wv : Wk;
    wb[idx] = (_Float16)W[K * DG + n];
}

__global__ __launch_bounds__(256, 1)
void main_kernel(const float* __restrict__ tokens, const float* __restrict__ qs,
                 const _Float16* __restrict__ wb,
                 float* __restrict__ gnum, float* __restrict__ gden) {
    __shared__ float qsl[DG];                 // q (scale folded), this batch
    __shared__ float se[MB * NHEADS];         // sim, then e = exp(sim)
    __shared__ float numer[DG];               // per-block numerator partials

    const int tid  = threadIdx.x;
    const int w    = tid >> 5;                // wave 0..7  (== head index for this wave)
    const int lane = tid & 31;
    const long R0  = (long)blockIdx.x * MB;   // first token row (128 | 8192 -> single batch)
    const int  b   = (int)(R0 / SEQ);

    qsl[tid]         = qs[b * DG + tid];
    qsl[tid + 256]   = qs[b * DG + tid + 256];
    numer[tid]       = 0.0f;
    numer[tid + 256] = 0.0f;
    for (int i = tid; i < MB * NHEADS; i += 256) se[i] = 0.0f;
    __syncthreads();

    const int  laneM = lane & 15;             // A row in tile / C column in tile
    const int  c0    = (lane < 16) ? 0 : 8;   // A-frag K sub-offset (ISA 16-bit A layout)
    const int  Mofs  = (lane < 16) ? 0 : 8;   // C layout: VGPR i -> row i or 8+i
    const float* tokRow0 = tokens + (R0 + laneM) * (long)DT + c0;
    const v16h* bp = (const v16h*)wb;

    auto gemm = [&](int m, int ct0, v8f (&acc)[2][ROWT]) {
        for (int kt = 0; kt < KT; ++kt) {
            v16h b0 = bp[((m * NT + ct0)     * KT + kt) * 32 + lane];
            v16h b1 = bp[((m * NT + ct0 + 1) * KT + kt) * 32 + lane];
            const float* tk = tokRow0 + kt * 32;
#pragma unroll
            for (int rt = 0; rt < ROWT; ++rt) {
                const float* ap = tk + (long)rt * 16 * DT;
                float4 f0 = *(const float4*)(ap);        // K = k0+c0 .. +3
                float4 f1 = *(const float4*)(ap + 4);    // K = k0+c0+4 .. +7
                float4 f2 = *(const float4*)(ap + 16);   // K = k0+c0+16 .. +19
                float4 f3 = *(const float4*)(ap + 20);   // K = k0+c0+20 .. +23
                v16h a;
                a[0]  = (_Float16)f0.x; a[1]  = (_Float16)f0.y;
                a[2]  = (_Float16)f0.z; a[3]  = (_Float16)f0.w;
                a[4]  = (_Float16)f1.x; a[5]  = (_Float16)f1.y;
                a[6]  = (_Float16)f1.z; a[7]  = (_Float16)f1.w;
                a[8]  = (_Float16)f2.x; a[9]  = (_Float16)f2.y;
                a[10] = (_Float16)f2.z; a[11] = (_Float16)f2.w;
                a[12] = (_Float16)f3.x; a[13] = (_Float16)f3.y;
                a[14] = (_Float16)f3.z; a[15] = (_Float16)f3.w;
                acc[0][rt] = __builtin_amdgcn_wmma_f32_16x16x32_f16(
                    false, a, false, b0, (short)0, acc[0][rt], false, false);
                acc[1][rt] = __builtin_amdgcn_wmma_f32_16x16x32_f16(
                    false, a, false, b1, (short)0, acc[1][rt], false, false);
            }
        }
    };

    // ---------- Stage 1: k = tanh(X @ Wk); sim[r][h] += q . k ----------
    // Wave w exclusively owns head w: its sim slots touch 8 private LDS banks,
    // and ds_add_f32 (no return) never stalls the wave -> conflicts fully hidden.
    for (int p = 0; p < 2; ++p) {
        int ct0 = w * 4 + p * 2;
        v8f acc[2][ROWT] = {};
        gemm(0, ct0, acc);
        float qv0 = qsl[ ct0      * 16 + laneM];        // q weight for this lane's d
        float qv1 = qsl[(ct0 + 1) * 16 + laneM];
#pragma unroll
        for (int rt = 0; rt < ROWT; ++rt)
#pragma unroll
            for (int i = 0; i < 8; ++i) {
                int Ml = rt * 16 + i + Mofs;            // local row 0..127
                float c = qv0 * fast_tanh(acc[0][rt][i])
                        + qv1 * fast_tanh(acc[1][rt][i]);   // fuse both j-tiles
                atomicAdd(&se[Ml * NHEADS + w], c);
            }
    }
    __syncthreads();

    // exp (|sim| <= 8, so no max-subtraction needed), then per-head denominators
    for (int i = tid; i < MB * NHEADS; i += 256) se[i] = __expf(se[i]);
    __syncthreads();
    if (tid < NHEADS) {
        float den = 0.0f;
        for (int r = 0; r < MB; ++r) den += se[r * NHEADS + tid];
        atomicAdd(&gden[b * NHEADS + tid], den);
    }

    // ---------- Stage 2: v = gelu(X @ Wv); numer[n] += sum_r e[r][h] * v ----------
    for (int p = 0; p < 2; ++p) {
        int ct0 = w * 4 + p * 2;
        v8f acc[2][ROWT] = {};
        gemm(1, ct0, acc);
        float s0 = 0.0f, s1 = 0.0f;
#pragma unroll
        for (int rt = 0; rt < ROWT; ++rt)
#pragma unroll
            for (int i = 0; i < 8; ++i) {
                int Ml = rt * 16 + i + Mofs;
                float ev = se[Ml * NHEADS + w];         // one LDS read feeds both tiles
                s0 += ev * gelu_erf(acc[0][rt][i]);
                s1 += ev * gelu_erf(acc[1][rt][i]);
            }
        atomicAdd(&numer[ ct0      * 16 + laneM], s0);  // 2-way conflict max
        atomicAdd(&numer[(ct0 + 1) * 16 + laneM], s1);
    }
    __syncthreads();

    atomicAdd(&gnum[b * DG + tid],       numer[tid]);
    atomicAdd(&gnum[b * DG + tid + 256], numer[tid + 256]);
}

__global__ void fin_kernel(const float* __restrict__ gnum, const float* __restrict__ gden,
                           float* __restrict__ out) {
    int idx = blockIdx.x * 256 + threadIdx.x;           // 0..4095
    int b = idx >> 9, n = idx & (DG - 1);
    out[idx] = gnum[idx] / gden[b * NHEADS + (n >> 6)];
}

extern "C" void kernel_launch(void* const* d_in, const int* in_sizes, int n_in,
                              void* d_out, int out_size, void* d_ws, size_t ws_size,
                              hipStream_t stream) {
    const float* tokens = (const float*)d_in[0];
    const float* ann    = (const float*)d_in[1];
    const float* Wq     = (const float*)d_in[2];
    const float* Wk     = (const float*)d_in[3];
    const float* Wv     = (const float*)d_in[4];

    char* ws = (char*)d_ws;
    float*    qsd  = (float*)(ws + WS_QS);
    float*    gnum = (float*)(ws + WS_GNUM);
    float*    gden = (float*)(ws + WS_GDEN);
    _Float16* wbuf = (_Float16*)(ws + WS_WB);

    zero_kernel<<<16, 256, 0, stream>>>(gnum, gden);
    qs_kernel<<<16, 256, 0, stream>>>(ann, Wq, qsd);
    swz_kernel<<<4096, 256, 0, stream>>>(Wk, Wv, wbuf);
    main_kernel<<<(BATCH * SEQ) / MB, 256, 0, stream>>>(tokens, qsd, wbuf, gnum, gden);
    fin_kernel<<<16, 256, 0, stream>>>(gnum, gden, (float*)d_out);
}
